// HighFrequencyDFT_85066122265290
// MI455X (gfx1250) — matrix operand order repaired and checked by reading, not compiled, not run on hardware
//
#include <hip/hip_runtime.h>
#include <hip/hip_bf16.h>

// CDNA5 / gfx1250: wave32, WMMA f32_16x16x32_f16 path.
typedef __attribute__((ext_vector_type(16))) _Float16 v16h;
typedef __attribute__((ext_vector_type(8)))  float    v8f;

#define MAT_H   160
#define MAT_W   160
#define SCALE   8
#define PATCH   32
#define GDIM    1304                 // (160-1)*8 + 32
#define NPATCH  (MAT_H * MAT_W)
#define WPB     8                    // waves per block (256 threads)
#define LPITCH  40                   // LDS row pitch in halves (80B -> 16B aligned rows)
#define NTW     10                   // twiddle tiles: Ca0,Ca1,Sa0,Sa1,Cb0,Cb1,Sb0,Sb1,Nb0,Nb1
#define TWELEM  (NTW * 32 * 16)      // halves: tile * lane * j

__device__ __forceinline__ void twiddle32(int a, int b, float& c, float& s) {
    int ph = (a * b) & 31;                        // exact mod-32 phase
    float ang = (float)ph * 0.19634954084936207f; // 2*pi/32
    c = __cosf(ang);
    s = __sinf(ang);
}

// ---------------------------------------------------------------- grayscale
__global__ void gray_kernel(const float* __restrict__ x,
                            const float* __restrict__ w,
                            float* __restrict__ gray, int n) {
    int i = blockIdx.x * blockDim.x + threadIdx.x;
    if (i < n) {
        float w0 = w[0], w1 = w[1], w2 = w[2];
        gray[i] = fmaf(x[i], w0, fmaf(x[i + n], w1, x[i + 2 * n] * w2));
    }
}

// ------------------------------------------- twiddle precompute (runs once per call)
// Materializes the 10 DFT twiddle tiles in exact per-lane WMMA register layout:
//   tw[t*512 + lane*16 + j]  ->  wave loads v16h with two global_load_b128, L2-resident.
// A-layout (16x32 f16): lane -> M = lm (+16*ti); element j -> K = j + 16*hi
// B-layout (32x16 f16): lane -> N = lm (+16*tj); element j -> K = j + 16*hi
__global__ void twiddle_kernel(_Float16* __restrict__ tw, unsigned int* __restrict__ gmax) {
    int idx = blockIdx.x * blockDim.x + threadIdx.x;
    if (idx == 0) *gmax = 0u;
    if (idx >= TWELEM) return;
    int t    = idx >> 9;          // tile 0..9
    int r    = idx & 511;
    int lane = r >> 4;
    int j    = r & 15;
    int hi = lane >> 4, lm = lane & 15;
    float val;
    if (t < 4) {                  // A-operand tiles: Ca[0],Ca[1],Sa[0],Sa[1]
        int ti = t & 1;
        int M = ti * 16 + lm;
        int K = j + 16 * hi;
        float c, s; twiddle32(M, K, c, s);
        val = (t < 2) ? c : s;
    } else {                      // B-operand tiles: Cb[0],Cb[1],Sb[0],Sb[1],Nb[0],Nb[1]
        int q  = t - 4;
        int tj = q & 1;
        int N = tj * 16 + lm;
        int K = j + 16 * hi;
        float c, s; twiddle32(K, N, c, s);
        val = (q < 2) ? c : ((q < 4) ? s : -s);
    }
    tw[idx] = (_Float16)val;
}

// ------------------------------------------------- per-patch FFT-as-GEMM
// One wave32 per patch. F = D P D with D = C - iS (32-pt DFT, symmetric).
// Stage 1: T1 = C*P, T2 = S*P      (A = twiddle tiles, B = patch tiles)
// Stage 2: Re = T1*C + T2*(-S),  Im = T1*S + T2*C   (sign of Im irrelevant)
__global__ __launch_bounds__(256) void patch_fft_kernel(
        const float* __restrict__ gray,
        const _Float16* __restrict__ tw,
        float* __restrict__ mat,
        unsigned int* __restrict__ gmax) {
    // [wave][buf][row][col] halves; pitch 80B keeps 16B alignment for merges
    __shared__ _Float16 lds[WPB][2][PATCH][LPITCH];

    const int wid  = threadIdx.x >> 5;
    const int lane = threadIdx.x & 31;
    const int hi   = lane >> 4;       // half-wave select
    const int lm   = lane & 15;

    const int pid = blockIdx.x * WPB + wid;     // grid sized exactly: always valid
    const int pi  = pid / MAT_W;
    const int pj  = pid % MAT_W;

    // ---- twiddle tiles: straight v16h loads (32B/lane, all L0/L2 hits)
    const v16h* twv = (const v16h*)tw;          // [tile*32 + lane]
    v16h Ca[2], Sa[2], Cb[2], Sb[2], Nb[2];
    Ca[0] = twv[0 * 32 + lane];
    Ca[1] = twv[1 * 32 + lane];
    Sa[0] = twv[2 * 32 + lane];
    Sa[1] = twv[3 * 32 + lane];
    Cb[0] = twv[4 * 32 + lane];
    Cb[1] = twv[5 * 32 + lane];
    Sb[0] = twv[6 * 32 + lane];
    Sb[1] = twv[7 * 32 + lane];
    Nb[0] = twv[8 * 32 + lane];
    Nb[1] = twv[9 * 32 + lane];

    // ---- stage patch into LDS, TRANSPOSED: lds[..][0][n][m] = P[m][n]
    // lane loads one full 32-float row of the patch (128B, float4-vectorized)
    {
        const float* src = gray + (size_t)(pi * SCALE + lane) * GDIM + pj * SCALE;
        float rowv[PATCH];
#pragma unroll
        for (int q = 0; q < 8; ++q) {
            float4 f = reinterpret_cast<const float4*>(src)[q];
            rowv[4 * q + 0] = f.x; rowv[4 * q + 1] = f.y;
            rowv[4 * q + 2] = f.z; rowv[4 * q + 3] = f.w;
        }
#pragma unroll
        for (int n = 0; n < PATCH; ++n)
            lds[wid][0][n][lane] = (_Float16)rowv[n];
    }
    __syncthreads();

    // ---- patch as B operand: Pb[tj][j] = P[16*hi + j][tj*16 + lm]
    v16h Pb[2];
#pragma unroll
    for (int tj = 0; tj < 2; ++tj) {
        int n = tj * 16 + lm;
#pragma unroll
        for (int j = 0; j < 16; ++j)
            Pb[tj][j] = lds[wid][0][n][16 * hi + j];   // contiguous -> ds_load_b128
    }

    // ---- stage 1: T1 = C*P, T2 = S*P  (8 WMMAs, f32 accum)
    v8f T1[2][2], T2[2][2];
#pragma unroll
    for (int ti = 0; ti < 2; ++ti) {
#pragma unroll
        for (int tj = 0; tj < 2; ++tj) {
            v8f z = {};
            T1[ti][tj] = __builtin_amdgcn_wmma_f32_16x16x32_f16(
                false, Ca[ti], false, Pb[tj], (short)0, z, false, false);
            T2[ti][tj] = __builtin_amdgcn_wmma_f32_16x16x32_f16(
                false, Sa[ti], false, Pb[tj], (short)0, z, false, false);
        }
    }
    __syncthreads();

    // ---- relayout T1,T2 (C/D layout) -> A layout via LDS (as f16)
    // C/D layout: VGPR k -> row u = k + 8*hi, col v = lm (within tile)
#pragma unroll
    for (int ti = 0; ti < 2; ++ti) {
#pragma unroll
        for (int tj = 0; tj < 2; ++tj) {
#pragma unroll
            for (int k = 0; k < 8; ++k) {
                int u = ti * 16 + k + 8 * hi;
                int v = tj * 16 + lm;
                lds[wid][0][u][v] = (_Float16)T1[ti][tj][k];
                lds[wid][1][u][v] = (_Float16)T2[ti][tj][k];
            }
        }
    }
    __syncthreads();

    v16h T1a[2], T2a[2];
#pragma unroll
    for (int ti = 0; ti < 2; ++ti) {
        int M = ti * 16 + lm;
#pragma unroll
        for (int j = 0; j < 16; ++j) {
            int K = j + 16 * hi;  // one 32B-contiguous run
            T1a[ti][j] = lds[wid][0][M][K];
            T2a[ti][j] = lds[wid][1][M][K];
        }
    }

    // ---- stage 2 (16 WMMAs) + log-power + masked high-frequency sum
    float sum = 0.0f;
#pragma unroll
    for (int ti = 0; ti < 2; ++ti) {
#pragma unroll
        for (int tj = 0; tj < 2; ++tj) {
            v8f z = {};
            v8f re = __builtin_amdgcn_wmma_f32_16x16x32_f16(
                false, T2a[ti], false, Nb[tj], (short)0, z, false, false);
            re = __builtin_amdgcn_wmma_f32_16x16x32_f16(
                false, T1a[ti], false, Cb[tj], (short)0, re, false, false);
            v8f im = __builtin_amdgcn_wmma_f32_16x16x32_f16(
                false, T2a[ti], false, Cb[tj], (short)0, z, false, false);
            im = __builtin_amdgcn_wmma_f32_16x16x32_f16(
                false, T1a[ti], false, Sb[tj], (short)0, im, false, false);

            int v = tj * 16 + lm;
            bool vin = (v >= 7) && (v <= 24);
#pragma unroll
            for (int k = 0; k < 8; ++k) {
                int u = ti * 16 + k + 8 * hi;
                bool take = ((u >= 7) && (u <= 24)) || vin;
                float p = fmaf(re[k], re[k], im[k] * im[k]);
                if (take) sum += __logf(p + 1.0f);
            }
        }
    }

    // ---- wave32 reduction, then write + global max (bit-pattern max: sum >= 0)
#pragma unroll
    for (int off = 16; off >= 1; off >>= 1)
        sum += __shfl_xor(sum, off, 32);
    if (lane == 0) {
        mat[pid] = sum;
        atomicMax(gmax, __float_as_uint(sum));
    }
}

// ---------------------------------------------------------------- normalize
__global__ void norm_kernel(const float* __restrict__ mat,
                            const unsigned int* __restrict__ gmax,
                            float* __restrict__ out, int n) {
    int i = blockIdx.x * blockDim.x + threadIdx.x;
    if (i < n) out[i] = mat[i] / __uint_as_float(*gmax);
}

// ------------------------------------------------------------------ launch
extern "C" void kernel_launch(void* const* d_in, const int* in_sizes, int n_in,
                              void* d_out, int out_size, void* d_ws, size_t ws_size,
                              hipStream_t stream) {
    const float* x    = (const float*)d_in[0];   // (1,3,1304,1304) f32
    const float* rgbw = (const float*)d_in[1];   // 3 f32
    // d_in[2], d_in[3] are mat_h/mat_w scalars (160,160) — statically known here.

    float*        gray = (float*)d_ws;                           // 1304*1304 f32
    float*        mat  = gray + (size_t)GDIM * GDIM;             // 25600 f32
    unsigned int* gmax = (unsigned int*)(mat + NPATCH);          // 1 u32
    _Float16*     tw   = (_Float16*)(mat + NPATCH + 16);         // 5120 f16, 32B-aligned

    const int npix = GDIM * GDIM;
    gray_kernel<<<(npix + 255) / 256, 256, 0, stream>>>(x, rgbw, gray, npix);
    twiddle_kernel<<<(TWELEM + 255) / 256, 256, 0, stream>>>(tw, gmax);
    patch_fft_kernel<<<NPATCH / WPB, 32 * WPB, 0, stream>>>(gray, tw, mat, gmax);
    norm_kernel<<<(NPATCH + 255) / 256, 256, 0, stream>>>(mat, gmax, (float*)d_out, NPATCH);
}
